// PureCartesianSparseE3Conv_66881230733944
// MI455X (gfx1250) — matrix operand order repaired and checked by reading, not compiled, not run on hardware
//
#include <hip/hip_runtime.h>

typedef __attribute__((ext_vector_type(16))) _Float16 v16h;
typedef __attribute__((ext_vector_type(8)))  _Float16 v8h;
typedef __attribute__((ext_vector_type(8)))  float    v8f;

#define KPAD 544   // 17 * 32 (512 h(x)d rows + 8 bias rows + 24 zero pad)
#define NKT  17
#define NNT  12    // 192 output cols = 3 paths * 8 u * 8 w

__device__ __forceinline__ float siluf(float x) {
  return x / (1.0f + __expf(-x));
}

union V16U { v16h v; struct { v8h lo; v8h hi; } h; };

// A-operand (M=16 x K=32, f16): lane L holds row M=L%16; K chunk depends on L/16.
// K(i) = kt*32 + 16*(L/16) + i        for i in [0,16)
__device__ __forceinline__ v16h load_a_tile(const _Float16* zrow, int kt, int H) {
  V16U u;
  u.h.lo = *(const v8h*)(zrow + kt * 32 + 16 * H);
  u.h.hi = *(const v8h*)(zrow + kt * 32 + 16 * H + 8);
  return u.v;
}

// B-operand tiles are pre-swizzled: tile t, lane L -> 16 contiguous f16.
__device__ __forceinline__ v16h load_b_tile(const _Float16* p) {
  V16U u;
  u.h.lo = *(const v8h*)(p);
  u.h.hi = *(const v8h*)(p + 8);
  return u.v;
}

// ---------------------------------------------------------------------------
// Zero output + in-degree counters
__global__ void init_kernel(float* __restrict__ out, size_t out_n,
                            int* __restrict__ cnt, int N) {
  size_t i = (size_t)blockIdx.x * blockDim.x + threadIdx.x;
  size_t stride = (size_t)gridDim.x * blockDim.x;
  for (size_t k = i; k < out_n; k += stride) out[k] = 0.0f;
  for (size_t k = i; k < (size_t)N; k += stride) cnt[k] = 0;
}

// ---------------------------------------------------------------------------
// In-degree histogram
__global__ void count_kernel(const int* __restrict__ edst, int* __restrict__ cnt, int E) {
  int e = blockIdx.x * blockDim.x + threadIdx.x;
  if (e < E) atomicAdd(&cnt[edst[e]], 1);
}

// ---------------------------------------------------------------------------
// Node scalar features: Ai = silu(emb[A] @ mlp_w1 + b1) @ mlp_w2 + b2   (N x 8)
__global__ void node_kernel(const float* __restrict__ emb, const int* __restrict__ A,
                            const float* __restrict__ w1, const float* __restrict__ b1,
                            const float* __restrict__ w2, const float* __restrict__ b2,
                            float* __restrict__ Ai, int N) {
  int n = blockIdx.x * blockDim.x + threadIdx.x;
  if (n >= N) return;
  float x[16];
  int a = A[n];
#pragma unroll
  for (int i = 0; i < 16; ++i) x[i] = emb[a * 16 + i];
  float acc[8];
#pragma unroll
  for (int w = 0; w < 8; ++w) acc[w] = b2[w];
  for (int k = 0; k < 64; ++k) {
    float t = b1[k];
#pragma unroll
    for (int i = 0; i < 16; ++i) t += x[i] * w1[i * 64 + k];
    t = siluf(t);
#pragma unroll
    for (int w = 0; w < 8; ++w) acc[w] += t * w2[k * 8 + w];
  }
#pragma unroll
  for (int w = 0; w < 8; ++w) Ai[(size_t)n * 8 + w] = acc[w];
}

// ---------------------------------------------------------------------------
// Pre-swizzle weights into WMMA B-operand layout (f16), one thread per (tile,lane).
//   Bmain: [17 kt][12 nt] tiles of K=32 x N=16, rows = (k*8+v | 512+v | pad),
//          cols = p*64+u*8+w, source fc_w3[k,2560]/fc_b3.
//   Bw1  : 4 tiles (K=16 padded to 32, N=64) from fc_w1
//   Bw2  : 8 tiles (K=64, N=64) from fc_w2
__global__ void prep_kernel(const float* __restrict__ fcw1, const float* __restrict__ fcw2,
                            const float* __restrict__ fcw3, const float* __restrict__ fcb3,
                            _Float16* __restrict__ Bmain, _Float16* __restrict__ Bw1,
                            _Float16* __restrict__ Bw2) {
  int idx = blockIdx.x * blockDim.x + threadIdx.x;
  if (idx >= 216 * 32) return;
  int t = idx >> 5, L = idx & 31;
  int half = L >> 4, lcol = L & 15;
  if (t < 204) {
    int kt = t / 12, nt = t % 12;
    int colp = nt * 16 + lcol;                       // p*64 + u*8 + w
    int p = colp >> 6, u = (colp >> 3) & 7, w = colp & 7;
    int scol = p * 512 + u * 64 + w;
#pragma unroll
    for (int i = 0; i < 16; ++i) {
      int krow = kt * 32 + half * 16 + i;
      float val;
      if (krow < 512)      val = fcw3[(size_t)(krow >> 3) * 2560 + scol + (krow & 7) * 8];
      else if (krow < 520) val = fcb3[scol + (krow - 512) * 8];
      else                 val = 0.0f;
      Bmain[(size_t)t * 512 + L * 16 + i] = (_Float16)val;
    }
  } else if (t < 208) {
    int nt = t - 204;
    int col = nt * 16 + lcol;
#pragma unroll
    for (int i = 0; i < 16; ++i) {
      int krow = half * 16 + i;
      float val = (krow < 16) ? fcw1[krow * 64 + col] : 0.0f;
      Bw1[(size_t)nt * 512 + L * 16 + i] = (_Float16)val;
    }
  } else {
    int tt = t - 208;
    int kt = tt >> 2, nt = tt & 3;
    int col = nt * 16 + lcol;
#pragma unroll
    for (int i = 0; i < 16; ++i) {
      int krow = kt * 32 + half * 16 + i;
      Bw2[(size_t)tt * 512 + L * 16 + i] = (_Float16)fcw2[krow * 64 + col];
    }
  }
}

// ---------------------------------------------------------------------------
// Main fused edge kernel: one wave32 per 16-edge tile.
__global__ __launch_bounds__(32)
void edge_kernel(const float* __restrict__ pos, const int* __restrict__ esrc,
                 const int* __restrict__ edst, const float* __restrict__ shifts,
                 const float* __restrict__ cell, const int* __restrict__ batch,
                 const float* __restrict__ Ai,
                 const float* __restrict__ fb1, const float* __restrict__ fb2,
                 const _Float16* __restrict__ Bw1, const _Float16* __restrict__ Bw2,
                 const _Float16* __restrict__ Bmain,
                 float* __restrict__ out, int E) {
  __shared__ __align__(16) _Float16 z[16][KPAD];   // A operand rows (per edge)
  __shared__ float h2s[16][64];
  __shared__ float ss[16][8];
  __shared__ float ds[16][8];
  __shared__ float ns[16][4];
  __shared__ float gs[16][24];
  __shared__ int   dsts[16];

  const int L = threadIdx.x;
  const int el = L & 15;
  const int H = L >> 4;
  const int e = blockIdx.x * 16 + el;
  const bool valid = (e < E);

  // ---- phase 1: geometry + gaussian basis (half-wave 0), s/d gather (half-wave 1)
  if (H == 0) {
    if (valid) {
      int s = esrc[e], d = edst[e], b = batch[s];
      float s0 = shifts[(size_t)e * 3 + 0], s1 = shifts[(size_t)e * 3 + 1],
            s2 = shifts[(size_t)e * 3 + 2];
      float ev[3]; float r2 = 0.0f;
#pragma unroll
      for (int j = 0; j < 3; ++j) {
        float sh = s0 * cell[b * 9 + j] + s1 * cell[b * 9 + 3 + j] + s2 * cell[b * 9 + 6 + j];
        ev[j] = pos[(size_t)d * 3 + j] - pos[(size_t)s * 3 + j] + sh;
        r2 += ev[j] * ev[j];
      }
      float r = sqrtf(r2);
      float invr = 1.0f / r;
      ns[el][0] = ev[0] * invr; ns[el][1] = ev[1] * invr; ns[el][2] = ev[2] * invr;
      dsts[el] = d;
      // soft_one_hot gaussian, *sqrt(16)/1.12
      v8h e0, e1, zz = {};
#pragma unroll
      for (int i = 0; i < 8; ++i) {
        float d0 = (r - 5.0f * (float)(i + 1) / 17.0f) * 3.4f;
        float d1 = (r - 5.0f * (float)(i + 9) / 17.0f) * 3.4f;
        e0[i] = (_Float16)(__expf(-d0 * d0) * 3.5714286f);
        e1[i] = (_Float16)(__expf(-d1 * d1) * 3.5714286f);
      }
      *(v8h*)&z[el][0]  = e0;
      *(v8h*)&z[el][8]  = e1;
      *(v8h*)&z[el][16] = zz;       // K padding for GEMM1
      *(v8h*)&z[el][24] = zz;
    } else {
      v8h zz = {};
      *(v8h*)&z[el][0] = zz; *(v8h*)&z[el][8] = zz;
      *(v8h*)&z[el][16] = zz; *(v8h*)&z[el][24] = zz;
      ns[el][0] = ns[el][1] = ns[el][2] = 0.0f;
      dsts[el] = -1;
    }
  } else {
    if (valid) {
      int s = esrc[e], d = edst[e];
#pragma unroll
      for (int u = 0; u < 8; ++u) {
        ss[el][u] = Ai[(size_t)s * 8 + u];
        ds[el][u] = Ai[(size_t)d * 8 + u];
      }
    } else {
#pragma unroll
      for (int u = 0; u < 8; ++u) { ss[el][u] = 0.0f; ds[el][u] = 0.0f; }
    }
  }
  __syncthreads();

  // ---- GEMM1: H1 = silu(EMB(16x32) @ W1 + b1), N=64
  {
    v16h a = load_a_tile(&z[el][0], 0, H);
    v8f acc1[4];
#pragma unroll
    for (int nt = 0; nt < 4; ++nt) {
      v16h b = load_b_tile(Bw1 + (size_t)nt * 512 + L * 16);
      v8f c = {};
      acc1[nt] = __builtin_amdgcn_wmma_f32_16x16x32_f16(false, a, false, b,
                                                        (short)0, c, false, false);
    }
    __syncthreads();   // A reads done before overwriting z rows
#pragma unroll
    for (int nt = 0; nt < 4; ++nt) {
      int col = nt * 16 + el;
      float bb = fb1[col];
#pragma unroll
      for (int j = 0; j < 8; ++j)
        z[j + 8 * H][col] = (_Float16)siluf(acc1[nt][j] + bb);
    }
  }
  __syncthreads();

  // ---- GEMM2: H2 = silu(H1(16x64) @ W2 + b2), N=64
  {
    v8f acc2[4] = {};
#pragma unroll
    for (int kt = 0; kt < 2; ++kt) {
      v16h a = load_a_tile(&z[el][0], kt, H);
#pragma unroll
      for (int nt = 0; nt < 4; ++nt) {
        v16h b = load_b_tile(Bw2 + (size_t)(kt * 4 + nt) * 512 + L * 16);
        acc2[nt] = __builtin_amdgcn_wmma_f32_16x16x32_f16(false, a, false, b,
                                                          (short)0, acc2[nt], false, false);
      }
    }
    __syncthreads();
#pragma unroll
    for (int nt = 0; nt < 4; ++nt) {
      int col = nt * 16 + el;
      float bb = fb2[col];
#pragma unroll
      for (int j = 0; j < 8; ++j)
        h2s[j + 8 * H][col] = siluf(acc2[nt][j] + bb);
    }
  }
  __syncthreads();

  // ---- form z = h2 (x) d  (rows k*8+v), bias rows 512..519 = d, pad 520..543 = 0
  {
    float dv[8];
#pragma unroll
    for (int v = 0; v < 8; ++v) dv[v] = ds[el][v];
    for (int kk = 0; kk < 32; ++kk) {
      int k = H * 32 + kk;
      float hk = h2s[el][k];
      v8h zz;
#pragma unroll
      for (int v = 0; v < 8; ++v) zz[v] = (_Float16)(hk * dv[v]);
      *(v8h*)&z[el][k * 8] = zz;
    }
    if (H == 0) {
      v8h dz, z0 = {};
#pragma unroll
      for (int v = 0; v < 8; ++v) dz[v] = (_Float16)dv[v];
      *(v8h*)&z[el][512] = dz;
      *(v8h*)&z[el][520] = z0;
      *(v8h*)&z[el][528] = z0;
      *(v8h*)&z[el][536] = z0;
    }
    for (int i = L; i < 16 * 24; i += 32) ((float*)gs)[i] = 0.0f;
  }
  __syncthreads();

  // ---- main GEMM: G(16x192) = Z(16x544) @ W3'(544x192), f32 accumulate
  v8f acc[NNT] = {};
  for (int kt = 0; kt < NKT; ++kt) {
    v16h a = load_a_tile(&z[el][0], kt, H);
    if (kt + 1 < NKT)
      __builtin_prefetch(Bmain + (size_t)(kt + 1) * NNT * 512 + L * 16, 0, 3);
#pragma unroll
    for (int nt = 0; nt < NNT; ++nt) {
      v16h b = load_b_tile(Bmain + (size_t)(kt * NNT + nt) * 512 + L * 16);
      acc[nt] = __builtin_amdgcn_wmma_f32_16x16x32_f16(false, a, false, b,
                                                       (short)0, acc[nt], false, false);
    }
  }

  // ---- epilogue: g[e][p*8+w] = (1/C) * sum_u s_u * G[e][p*64+u*8+w]
#pragma unroll
  for (int nt = 0; nt < NNT; ++nt) {
    int colp = nt * 16 + el;
    int p = colp >> 6, u = (colp >> 3) & 7, w = colp & 7;
#pragma unroll
    for (int j = 0; j < 8; ++j) {
      int er = j + 8 * H;
      atomicAdd(&gs[er][p * 8 + w], acc[nt][j] * ss[er][u] * 0.125f);
    }
  }
  __syncthreads();

  // ---- scatter: out0 (8) | g1 (x) n (24) | g2 (x) n (x) n (72)  -> 104 cols
  for (int idx = L; idx < 16 * 104; idx += 32) {
    int er = idx / 104, c = idx % 104;
    int d = dsts[er];
    if (d < 0) continue;
    float val;
    if (c < 8) {
      val = gs[er][c];
    } else if (c < 32) {
      int t = c - 8;
      val = gs[er][8 + t / 3] * ns[er][t % 3];
    } else {
      int t = c - 32;
      val = gs[er][16 + t / 9] * ns[er][(t % 9) / 3] * ns[er][t % 3];
    }
    atomicAdd(&out[(size_t)d * 208 + c], val);
  }
}

// ---------------------------------------------------------------------------
// Mean-normalize by clipped in-degree (cols >= 104 stay zero).
__global__ void norm_kernel(float* __restrict__ out, const int* __restrict__ cnt, int N) {
  int idx = blockIdx.x * blockDim.x + threadIdx.x;
  if (idx >= N * 104) return;
  int n = idx / 104, c = idx % 104;
  int cv = cnt[n]; if (cv < 1) cv = 1;
  out[(size_t)n * 208 + c] *= (1.0f / (float)cv);
}

// ---------------------------------------------------------------------------
extern "C" void kernel_launch(void* const* d_in, const int* in_sizes, int n_in,
                              void* d_out, int out_size, void* d_ws, size_t ws_size,
                              hipStream_t stream) {
  const float* pos    = (const float*)d_in[0];
  const int*   A      = (const int*)d_in[1];
  const int*   batch  = (const int*)d_in[2];
  const int*   esrc   = (const int*)d_in[3];
  const int*   edst   = (const int*)d_in[4];
  const float* shifts = (const float*)d_in[5];
  const float* cell   = (const float*)d_in[6];
  const float* emb    = (const float*)d_in[7];
  const float* mw1    = (const float*)d_in[8];
  const float* mb1    = (const float*)d_in[9];
  const float* mw2    = (const float*)d_in[10];
  const float* mb2    = (const float*)d_in[11];
  const float* fw1    = (const float*)d_in[12];
  const float* fb1    = (const float*)d_in[13];
  const float* fw2    = (const float*)d_in[14];
  const float* fb2    = (const float*)d_in[15];
  const float* fw3    = (const float*)d_in[16];
  const float* fb3    = (const float*)d_in[17];
  const int N = in_sizes[1];
  const int E = in_sizes[3];

  char* ws = (char*)d_ws;
  _Float16* Bmain = (_Float16*)ws;                    // 204 KiB (204 tiles * 1KB)
  _Float16* Bw1   = (_Float16*)(ws + 208896);         // 4 KiB
  _Float16* Bw2   = (_Float16*)(ws + 212992);         // 8 KiB
  float*    Ai    = (float*)(ws + 221184);            // N*8 f32
  int*      cnt   = (int*)(ws + 221184 + (size_t)N * 32);
  float*    out   = (float*)d_out;

  init_kernel<<<1024, 256, 0, stream>>>(out, (size_t)out_size, cnt, N);
  prep_kernel<<<(216 * 32 + 255) / 256, 256, 0, stream>>>(fw1, fw2, fw3, fb3,
                                                          Bmain, Bw1, Bw2);
  node_kernel<<<(N + 127) / 128, 128, 0, stream>>>(emb, A, mw1, mb1, mw2, mb2, Ai, N);
  count_kernel<<<(E + 255) / 256, 256, 0, stream>>>(edst, cnt, E);
  edge_kernel<<<(E + 15) / 16, 32, 0, stream>>>(pos, esrc, edst, shifts, cell, batch,
                                                Ai, fb1, fb2, Bw1, Bw2, Bmain, out, E);
  norm_kernel<<<(N * 104 + 255) / 256, 256, 0, stream>>>(out, cnt, N);
}
